// QuantizationLayer_41154376630529
// MI455X (gfx1250) — compile-verified, matrix operand order checked by reading, not run on hardware
//
#include <hip/hip_runtime.h>

// ---------------------------------------------------------------------------
// EST quantization layer for MI455X (gfx1250, wave32, WMMA).
//
// 9 bins x 800K events x (1->100->100->1 MLP) ~= 147 GFLOP dominated by the
// 100x100 layer -> batched [32 x 128] x [128 x 112] f16 WMMA GEMM tiles
// (f32 accumulate). Memory side ~67 MB (~3us @ 23.3 TB/s); the 24.9 MB voxel
// grid is L2-resident (192 MB), atomics stream to L2 atomic units.
//
// Structure: one wave owns one 32-event tile and loops its 9 temporal bins.
// w2^T is staged once per block (3,125 blocks total). M=32 per wave means each
// shared-B fragment load feeds two v_wmma ops (halves LDS BW per WMMA). The
// wave-private h1 tile needs no block barrier: a single `s_wait_dscnt 0`
// orders the packed ds_store_b128 staging against the fragment ds_load_b128s.
// ---------------------------------------------------------------------------

typedef __attribute__((ext_vector_type(16))) _Float16     v16h;
typedef __attribute__((ext_vector_type(8)))  float        v8f;
typedef __attribute__((ext_vector_type(4)))  unsigned int v4u;

#define C_BINS   9
#define H_DIM    180
#define W_DIM    240
#define B_DIM    8
#define N_EV     100000
#define NUM_VOX  (2 * C_BINS * H_DIM * W_DIM * B_DIM)   // 6,220,800
#define KP       128                                    // padded inner dim (100 -> 128)
#define NP       112                                    // padded hidden dim (100 -> 112)
#define TILES32  ((B_DIM * N_EV) / 32)                  // 25,000 (exact)
#define BLOCKS   (TILES32 / 8)                          // 3,125 (8 waves/block)

union Frag16 {
    struct { v4u lo, hi; } q;
    v16h h;
};

union Pack8h {
    v4u u;
    _Float16 h[8];
};

__device__ __forceinline__ float leaky01(float v) {
    return fmaxf(v, 0.f) + 0.1f * fminf(v, 0.f);
}

__global__ void zero_out_kernel(float4* __restrict__ out4, int n4) {
    int i = blockIdx.x * blockDim.x + threadIdx.x;
    int stride = gridDim.x * blockDim.x;
    for (; i < n4; i += stride) out4[i] = make_float4(0.f, 0.f, 0.f, 0.f);
}

__global__ __launch_bounds__(256) void est_kernel(
    const float* __restrict__ t,  const float* __restrict__ w1,
    const float* __restrict__ b1, const float* __restrict__ w2,
    const float* __restrict__ b2, const float* __restrict__ w3,
    const float* __restrict__ b3, const int* __restrict__ xs,
    const int* __restrict__ ys,   const int* __restrict__ ps,
    float* __restrict__ out) {

    // LDS: 28 KB w2^T f16 [n][k] + 64 KB h1 tiles (8 waves x 32x128 f16) + consts
    __shared__ __align__(16) _Float16 sW2T[NP * KP];
    __shared__ __align__(16) _Float16 sH1[8][32 * KP];
    __shared__ float sW1[KP], sB1[KP], sB2[NP], sW3[NP];

    const int tid = threadIdx.x;

    // ---- stage w2 transposed (+zero pad) as f16, k-major for coalesced reads
    for (int i = tid; i < NP * KP; i += 256) {
        int k = i / NP;
        int n = i - k * NP;
        float v = (k < 100 && n < 100) ? w2[k * 100 + n] : 0.f;
        sW2T[n * KP + k] = (_Float16)v;
    }
    if (tid < KP) {
        sW1[tid] = (tid < 100) ? w1[tid] : 0.f;
        sB1[tid] = (tid < 100) ? b1[tid] : 0.f;
    }
    if (tid < NP) {
        sB2[tid] = (tid < 100) ? b2[tid] : 0.f;
        sW3[tid] = (tid < 100) ? w3[tid] : 0.f;   // zero pad => pad cols add 0
    }
    __syncthreads();   // only block-wide barrier

    const int wave = tid >> 5;
    const int lane = tid & 31;
    const int m    = lane & 15;         // event row (per sub-tile) / N column
    const int half = lane >> 4;         // K-half selector per ISA 16-bit layout
    const int tile = blockIdx.x * 8 + wave;   // < 25,000, exact

    _Float16* sH1w = &sH1[wave][0];

    // ---- hoist per-event data out of the bin loop ----
    float tv[2];
    tv[0] = t[tile * 32 + m];
    tv[1] = t[tile * 32 + 16 + m];

    // scatter lanes: rowS in [0,16), lanes 0-7/16-23 active (8-15/24-31 dup)
    const int rowS   = (lane & 7) + half * 8;
    const bool activ = ((lane >> 3) & 1) == 0;
    int   idxb[2];
    float tsc[2];
    #pragma unroll
    for (int mt = 0; mt < 2; ++mt) {
        const int e  = tile * 32 + mt * 16 + rowS;
        const int bi = e / N_EV;
        idxb[mt] = xs[e] + W_DIM * ys[e]
                 + (W_DIM * H_DIM * C_BINS) * ps[e]
                 + (W_DIM * H_DIM * C_BINS * 2) * bi;
        tsc[mt]  = t[e];
    }
    const float b3v = b3[0];

    // ---- loop over the 9 temporal bins (rolled) ----
    for (int bin = 0; bin < C_BINS; ++bin) {
        const float binoff = (float)bin * 0.125f;

        // layer 1: h1[row][f] = leaky(s*w1[f]+b1[f]) -> packed ds_store_b128
        #pragma unroll
        for (int mt = 0; mt < 2; ++mt) {
            const float s0 = tv[mt] - binoff;
            #pragma unroll
            for (int j = 0; j < 8; ++j) {
                Pack8h pk;
                #pragma unroll
                for (int i = 0; i < 8; ++i) {
                    const int f = j * 16 + half * 8 + i;       // 0..127
                    pk.h[i] = (_Float16)leaky01(fmaf(s0, sW1[f], sB1[f]));
                }
                *(v4u*)&sH1w[(mt * 16 + m) * KP + j * 16 + half * 8] = pk.u;
            }
        }

        // wave-private tile: order LDS stores vs cross-lane fragment loads
        // with the CDNA5 split DS counter (no block barrier needed).
        asm volatile("s_wait_dscnt 0x0" ::: "memory");

        // A fragments for both 16-row sub-tiles: per ISA 16-bit A layout a
        // lane's 16 halves are two contiguous 8-half runs -> 2x ds_load_b128.
        Frag16 afrag[2][4];
        #pragma unroll
        for (int mt = 0; mt < 2; ++mt) {
            #pragma unroll
            for (int kt = 0; kt < 4; ++kt) {
                const _Float16* ap =
                    &sH1w[(mt * 16 + m) * KP + kt * 32 + half * 8];
                afrag[mt][kt].q.lo = *(const v4u*)(ap);
                afrag[mt][kt].q.hi = *(const v4u*)(ap + 16);
            }
        }

        // layer 2: [32 x 128] x [128 x 112]; each B fragment feeds 2 WMMAs.
        float dotp[2][8] = {};
        #pragma unroll
        for (int nt = 0; nt < 7; ++nt) {
            v8f acc0 = {};
            v8f acc1 = {};
            #pragma unroll
            for (int kt = 0; kt < 4; ++kt) {
                Frag16 bf;
                const _Float16* bp =
                    &sW2T[(nt * 16 + m) * KP + kt * 32 + half * 8];
                bf.q.lo = *(const v4u*)(bp);
                bf.q.hi = *(const v4u*)(bp + 16);
                acc0 = __builtin_amdgcn_wmma_f32_16x16x32_f16(
                    false, afrag[0][kt].h, false, bf.h, (short)0, acc0, false, false);
                acc1 = __builtin_amdgcn_wmma_f32_16x16x32_f16(
                    false, afrag[1][kt].h, false, bf.h, (short)0, acc1, false, false);
            }
            // epilogue: +b2, leaky, *w3 -> per-lane partial dot products.
            // C/D layout: lane -> col n = nt*16+m; VGPR r -> row r + 8*half.
            const float b2v = sB2[nt * 16 + m];
            const float w3v = sW3[nt * 16 + m];
            #pragma unroll
            for (int r = 0; r < 8; ++r) {
                dotp[0][r] = fmaf(leaky01(acc0[r] + b2v), w3v, dotp[0][r]);
                dotp[1][r] = fmaf(leaky01(acc1[r] + b2v), w3v, dotp[1][r]);
            }
        }

        // reduce over 16 column-lanes (xor masks < 16 stay within each half)
        #pragma unroll
        for (int mt = 0; mt < 2; ++mt) {
            #pragma unroll
            for (int r = 0; r < 8; ++r) {
                float v = dotp[mt][r];
                v += __shfl_xor(v, 1, 32);
                v += __shfl_xor(v, 2, 32);
                v += __shfl_xor(v, 4, 32);
                v += __shfl_xor(v, 8, 32);
                dotp[mt][r] = v;
            }
        }

        // scatter: lane picks its row's sum; one atomic per event per bin
        const int binofs = (W_DIM * H_DIM) * bin;
        #pragma unroll
        for (int mt = 0; mt < 2; ++mt) {
            float dsum = dotp[mt][0];
            #pragma unroll
            for (int r = 1; r < 8; ++r)
                if ((lane & 7) == r) dsum = dotp[mt][r];
            if (activ) {
                const float val = tsc[mt] * (dsum + b3v);
                int idx = idxb[mt] + binofs;
                idx = min(max(idx, 0), NUM_VOX - 1);
                atomicAdd(out + idx, val);
            }
        }
    }
}

extern "C" void kernel_launch(void* const* d_in, const int* in_sizes, int n_in,
                              void* d_out, int out_size, void* d_ws, size_t ws_size,
                              hipStream_t stream) {
    const float* t  = (const float*)d_in[0];
    const float* w1 = (const float*)d_in[1];
    const float* b1 = (const float*)d_in[2];
    const float* w2 = (const float*)d_in[3];
    const float* b2 = (const float*)d_in[4];
    const float* w3 = (const float*)d_in[5];
    const float* b3 = (const float*)d_in[6];
    const int*   xs = (const int*)d_in[7];
    const int*   ys = (const int*)d_in[8];
    const int*   ps = (const int*)d_in[9];
    float* out = (float*)d_out;

    // zero the voxel grid (out_size = 6,220,800 floats, divisible by 4)
    zero_out_kernel<<<4096, 256, 0, stream>>>((float4*)out, out_size / 4);

    // 25,000 wave-tiles (32 events x all 9 bins each), 8 waves per block
    est_kernel<<<BLOCKS, 256, 0, stream>>>(t, w1, b1, w2, b2, w3, b3,
                                           xs, ys, ps, out);
}
